// Model_25744033972364
// MI455X (gfx1250) — matrix operand order repaired and checked by reading, not compiled, not run on hardware
//
#include <hip/hip_runtime.h>
#include <hip/hip_bf16.h>

typedef __bf16 bf16;
typedef bf16 v16bf __attribute__((ext_vector_type(16)));
typedef bf16 v8bf  __attribute__((ext_vector_type(8)));
typedef float v8f  __attribute__((ext_vector_type(8)));
typedef uint32_t u32x4 __attribute__((ext_vector_type(4)));
typedef uint32_t u32x8 __attribute__((ext_vector_type(8)));

#define HID   64
#define DIN   300
#define DPAD  320                 // DIN padded to K-chunk multiple (10 x 32)
#define NLAY  3
#define BATCH 256
#define TSEQ  512
#define G4    256                 // 4*HID
#define MROWS (TSEQ * BATCH)      // 131072

__device__ __forceinline__ float sigmoidf_(float x) { return 1.0f / (1.0f + __expf(-x)); }

// low 32 bits of a flat generic address == wave-relative LDS byte offset (ISA 10.2)
__device__ __forceinline__ uint32_t lds_off(const void* p) {
  return (uint32_t)(uintptr_t)p;
}
// CDNA5 async copy: global -> LDS, 16B per lane, tracked with ASYNCcnt
__device__ __forceinline__ void async_b128(uint32_t lds, uint64_t gaddr) {
  asm volatile("global_load_async_to_lds_b128 %0, %1, off"
               :: "v"(lds), "v"(gaddr) : "memory");
}
#define ASYNC_WAIT_0() asm volatile("s_wait_asynccnt 0x0" ::: "memory")
#define ASYNC_WAIT_2() asm volatile("s_wait_asynccnt 0x2" ::: "memory")

// CDNA5 Tensor Data Mover: 1-D contiguous tile, global -> LDS (ISA ch.7/8 of
// async_tensor doc). D# group0 (4 SGPRs) + group1 (8 SGPRs); VADDR2/3 omitted
// (<=2D tensor). nelem in 4-byte units; one issuing wave moves the whole tile.
__device__ __forceinline__ void tdm_load_1d(uint32_t lds_byte, uint64_t gaddr,
                                            uint32_t nelem) {
  u32x4 g0;
  g0.x = 1u;                                          // count=1, user descriptor
  g0.y = lds_byte;                                    // lds_addr
  g0.z = (uint32_t)gaddr;                             // global_addr[31:0]
  g0.w = ((uint32_t)(gaddr >> 32) & 0x01FFFFFFu)      // global_addr[56:32]
       | (2u << 30);                                  // type=2 ("image")
  u32x8 g1;
  g1[0] = 2u << 16;                                   // data_size=2 (4 bytes)
  g1[1] = (nelem & 0xFFFFu) << 16;                    // tensor_dim0[15:0]
  g1[2] = (nelem >> 16) | (1u << 16);                 // tensor_dim0[31:16], tensor_dim1=1
  g1[3] = (nelem & 0xFFFFu) << 16;                    // tile_dim0
  g1[4] = 0;                                          // tile_dim1=0 (1D), tile_dim2=0
  g1[5] = nelem;                                      // tensor_dim0_stride[31:0]
  g1[6] = 0;
  g1[7] = 0;
  asm volatile("tensor_load_to_lds %0, %1" :: "s"(g0), "s"(g1) : "memory");
}

// ---- WMMA fragment loaders (wave32, layouts per CDNA5 ISA 7.12.2) ----------
__device__ __forceinline__ v16bf load_A_frag(const bf16* p, int stride) {
  const int lane = threadIdx.x & 31;
  const bf16* row = p + (lane & 15) * stride + ((lane >> 4) << 3);
  v8bf lo = *(const v8bf*)(row);
  v8bf hi = *(const v8bf*)(row + 16);
  return __builtin_shufflevector(lo, hi, 0,1,2,3,4,5,6,7,8,9,10,11,12,13,14,15);
}
__device__ __forceinline__ v16bf load_B_frag(const bf16* p, int stride) {
  const int lane = threadIdx.x & 31;
  const bf16* col = p + (lane & 15) * stride + ((lane >> 4) << 3);
  v8bf lo = *(const v8bf*)(col);
  v8bf hi = *(const v8bf*)(col + 16);
  return __builtin_shufflevector(lo, hi, 0,1,2,3,4,5,6,7,8,9,10,11,12,13,14,15);
}

// ---- one-time converters -----------------------------------------------------
__global__ __launch_bounds__(256)
void convert_x_kernel(const float* __restrict__ in, bf16* __restrict__ out) {
  size_t idx = (size_t)blockIdx.x * 256 + threadIdx.x;   // < MROWS*DPAD
  size_t row = idx / DPAD;
  int k = (int)(idx % DPAD);
  float v = 0.f;
  if (k < DIN) {
    size_t b = row & 255, t = row >> 8;
    v = in[(b * TSEQ + t) * DIN + k];
  }
  out[idx] = (bf16)v;
}
__global__ __launch_bounds__(256)
void convert_w_kernel(const float* __restrict__ in, bf16* __restrict__ out,
                      int N, int K, int Npad, int Kpad) {
  int idx = blockIdx.x * 256 + threadIdx.x;
  if (idx >= Npad * Kpad) return;
  int n = idx / Kpad, k = idx % Kpad;
  out[idx] = (bf16)((n < N && k < K) ? in[(size_t)n * K + k] : 0.f);
}

// ---- GEMM: C[M,N] = act( A[M,Kpad](bf16) * W[Npad,Kpad](bf16)^T + bias (+bias2) )
__global__ __launch_bounds__(256)
void gemm_bf16_wmma(const bf16* __restrict__ A, const bf16* __restrict__ W,
                    const float* __restrict__ bias, const float* __restrict__ bias2,
                    float* __restrict__ Cf, bf16* __restrict__ Cb,
                    int Kpad, int N, int ldc, int actTanh)
{
  __shared__ __align__(16) bf16 aT[2][64 * 40];
  __shared__ __align__(16) bf16 wT[2][64 * 40];
  const int tid = threadIdx.x;
  const int lane = tid & 31;
  const int wave = tid >> 5;
  const int rowTile = wave & 3;
  const int colPair = wave >> 2;
  const int m0 = blockIdx.x * 64;
  const int n0 = blockIdx.y * 64;
  const int r   = tid >> 2;
  const int seg = tid & 3;

  const bf16* gA = A + (size_t)(m0 + r) * Kpad + seg * 8;
  const bf16* gW = W + (size_t)(n0 + r) * Kpad + seg * 8;
  const uint32_t so = (uint32_t)(r * 80 + seg * 16);   // padded LDS stride 40 bf16
  uint32_t aB[2] = { lds_off(&aT[0][0]) + so, lds_off(&aT[1][0]) + so };
  uint32_t wB[2] = { lds_off(&wT[0][0]) + so, lds_off(&wT[1][0]) + so };

  v8f acc0 = {}, acc1 = {};
  const int nch = Kpad >> 5;

  async_b128(aB[0], (uint64_t)(uintptr_t)gA);
  async_b128(wB[0], (uint64_t)(uintptr_t)gW);
  for (int ci = 0; ci < nch; ++ci) {
    int cur = ci & 1;
    if (ci + 1 < nch) {
      async_b128(aB[cur ^ 1], (uint64_t)(uintptr_t)(gA + (ci + 1) * 32));
      async_b128(wB[cur ^ 1], (uint64_t)(uintptr_t)(gW + (ci + 1) * 32));
      ASYNC_WAIT_2();               // in-order: current chunk's pair is done
    } else {
      ASYNC_WAIT_0();
    }
    __syncthreads();
    v16bf aF  = load_A_frag(&aT[cur][rowTile * 16 * 40], 40);
    v16bf bF0 = load_B_frag(&wT[cur][(colPair * 32 +  0) * 40], 40);
    v16bf bF1 = load_B_frag(&wT[cur][(colPair * 32 + 16) * 40], 40);
    acc0 = __builtin_amdgcn_wmma_f32_16x16x32_bf16(false, aF, false, bF0, (short)0, acc0, false, false);
    acc1 = __builtin_amdgcn_wmma_f32_16x16x32_bf16(false, aF, false, bF1, (short)0, acc1, false, false);
    __syncthreads();                // buffer reuse distance = 2 iterations
  }

  const int mBase = m0 + rowTile * 16 + ((lane >> 4) << 3);
  const int nBase = n0 + colPair * 32 + (lane & 15);
  #pragma unroll
  for (int r8 = 0; r8 < 8; ++r8) {
    int m = mBase + r8;
    #pragma unroll
    for (int s = 0; s < 2; ++s) {
      int n = nBase + s * 16;
      float v = (s ? acc1[r8] : acc0[r8]);
      if (n < N) {
        v += bias[n] + (bias2 ? bias2[n] : 0.f);
        if (actTanh) v = tanhf(v);
        if (Cb) Cb[(size_t)m * ldc + n] = (bf16)v;
        else    Cf[(size_t)m * ldc + n] = v;
      }
    }
  }
}

// ---- Persistent LSTM layer: 8 blocks x 32 batch rows, loop over T -----------
// xp[t*B+b, 0:256] = x·Wih^T + bih + bhh (fp32). Gates += h·Whh^T via WMMA.
// Next step's contiguous 32KB xp tile is DMA'd into LDS by the Tensor Data
// Mover (one tensor_load_to_lds per step, issued by wave 0) while computing.
__global__ __launch_bounds__(256)
void lstm_layer_kernel(const float* __restrict__ xp, const float* __restrict__ whh,
                       const float* __restrict__ h0, const float* __restrict__ c0,
                       bf16* __restrict__ hs, float* __restrict__ hn, float* __restrict__ cn)
{
  __shared__ __align__(16) bf16 whhT[G4 * 72];
  __shared__ __align__(16) bf16 hT[32 * 72];
  __shared__ float cT[32 * HID];
  __shared__ float gT[32 * G4];
  __shared__ __align__(16) float xpS[2][32 * G4];   // TDM double buffer (2x32KB)
  const int tid  = threadIdx.x;
  const int lane = tid & 31;
  const int wave = tid >> 5;
  const int rowTile = wave & 1;
  const int gate    = wave >> 1;
  const int b0 = blockIdx.x * 32;

  {  // Whh fp32 -> bf16 LDS, one row per thread, vectorized
    const float* wr = whh + (size_t)tid * HID;
    #pragma unroll
    for (int u = 0; u < 8; ++u) {
      v8bf v;
      #pragma unroll
      for (int j = 0; j < 8; ++j) v[j] = (bf16)wr[u * 8 + j];
      *(v8bf*)&whhT[tid * 72 + u * 8] = v;
    }
  }
  {  // h0, c0
    int i0 = tid * 8, b = i0 >> 6, hh = i0 & 63;
    v8bf v;
    #pragma unroll
    for (int u = 0; u < 8; ++u) {
      v[u] = (bf16)h0[(size_t)b0 * HID + i0 + u];
      cT[i0 + u] = c0[(size_t)b0 * HID + i0 + u];
    }
    *(v8bf*)&hT[b * 72 + hh] = v;
  }

  const uint32_t xL[2] = { lds_off(&xpS[0][0]), lds_off(&xpS[1][0]) };
  if (wave == 0)  // preload t=0 tile via TDM (8192 floats, contiguous)
    tdm_load_1d(xL[0], (uint64_t)(uintptr_t)(xp + (size_t)b0 * G4), 32 * G4);

  const int mLoc  = rowTile * 16 + ((lane >> 4) << 3);
  const int nBase = gate * 64 + (lane & 15);

  for (int t = 0; t < TSEQ; ++t) {
    const int cur = t & 1;
    if (wave == 0) {
      if (t + 1 < TSEQ) {
        tdm_load_1d(xL[cur ^ 1],
                    (uint64_t)(uintptr_t)(xp + ((size_t)(t + 1) * BATCH + b0) * G4),
                    32 * G4);
        __builtin_amdgcn_s_wait_tensorcnt(1);  // tile t done; t+1 may be in flight
      } else {
        __builtin_amdgcn_s_wait_tensorcnt(0);
      }
    }
    __syncthreads();                // tile visible to all waves; also guards hT

    v8f acc[4];
    #pragma unroll
    for (int s = 0; s < 4; ++s)
      #pragma unroll
      for (int r = 0; r < 8; ++r)
        acc[s][r] = xpS[cur][(mLoc + r) * G4 + nBase + s * 16];

    #pragma unroll
    for (int kc = 0; kc < 2; ++kc) {
      v16bf aF = load_A_frag(hT + rowTile * 16 * 72 + kc * 32, 72);
      #pragma unroll
      for (int s = 0; s < 4; ++s) {
        v16bf bF = load_B_frag(whhT + (gate * 64 + s * 16) * 72 + kc * 32, 72);
        acc[s] = __builtin_amdgcn_wmma_f32_16x16x32_bf16(false, aF, false, bF, (short)0, acc[s], false, false);
      }
    }
    #pragma unroll
    for (int s = 0; s < 4; ++s)
      #pragma unroll
      for (int r = 0; r < 8; ++r)
        gT[(mLoc + r) * G4 + nBase + s * 16] = acc[s][r];
    __syncthreads();

    {  // fused gates: thread owns 8 contiguous hidden elems of one batch row
      int i0 = tid * 8, b = i0 >> 6, hh = i0 & 63;
      v8bf hv8;
      #pragma unroll
      for (int u = 0; u < 8; ++u) {
        float ig = sigmoidf_(gT[b * G4 +       hh + u]);
        float fg = sigmoidf_(gT[b * G4 +  64 + hh + u]);
        float gg = tanhf   (gT[b * G4 + 128 + hh + u]);
        float og = sigmoidf_(gT[b * G4 + 192 + hh + u]);
        float cv = fg * cT[i0 + u] + ig * gg;
        float hv = og * tanhf(cv);
        cT[i0 + u] = cv;
        hv8[u] = (bf16)hv;
      }
      *(v8bf*)&hT[b * 72 + hh] = hv8;
      *(v8bf*)(hs + ((size_t)t * BATCH + b0 + b) * HID + hh) = hv8;
    }
    __syncthreads();
  }
  {  // final h_n, c_n (fp32)
    int i0 = tid * 8, b = i0 >> 6, hh = i0 & 63;
    #pragma unroll
    for (int u = 0; u < 8; ++u) {
      hn[(size_t)b0 * HID + i0 + u] = (float)hT[b * 72 + hh + u];
      cn[(size_t)b0 * HID + i0 + u] = cT[i0 + u];
    }
  }
}

extern "C" void kernel_launch(void* const* d_in, const int* in_sizes, int n_in,
                              void* d_out, int out_size, void* d_ws, size_t ws_size,
                              hipStream_t stream) {
  const float* input = (const float*)d_in[0];
  const float* h0    = (const float*)d_in[1];
  const float* c0    = (const float*)d_in[2];
  const float* wih[NLAY], *whh[NLAY], *bih[NLAY], *bhh[NLAY];
  for (int l = 0; l < NLAY; ++l) {
    wih[l] = (const float*)d_in[3 + 4 * l];
    whh[l] = (const float*)d_in[4 + 4 * l];
    bih[l] = (const float*)d_in[5 + 4 * l];
    bhh[l] = (const float*)d_in[6 + 4 * l];
  }
  const float* w1 = (const float*)d_in[15];
  const float* b1 = (const float*)d_in[16];
  const float* w2 = (const float*)d_in[17];
  const float* b2 = (const float*)d_in[18];

  float* out = (float*)d_out;                       // [MROWS, DIN]
  float* hn  = out + (size_t)MROWS * DIN;
  float* cn  = hn + (size_t)NLAY * BATCH * HID;

  char* wsb = (char*)d_ws;
  size_t off = 0;
  auto alloc = [&](size_t bytes) {
    void* p = wsb + off;
    off = (off + bytes + 255) & ~(size_t)255;
    return p;
  };
  float* xpb  = (float*)alloc((size_t)MROWS * G4 * 4);   // gate precompute (fp32)
  bf16* xbf   = (bf16*) alloc((size_t)MROWS * DPAD * 2); // time-major input, bf16
  bf16* actA  = (bf16*) alloc((size_t)MROWS * HID * 2);
  bf16* actB  = (bf16*) alloc((size_t)MROWS * HID * 2);
  bf16* wih0b = (bf16*) alloc((size_t)G4 * DPAD * 2);
  bf16* wih1b = (bf16*) alloc((size_t)G4 * HID * 2);
  bf16* wih2b = (bf16*) alloc((size_t)G4 * HID * 2);
  bf16* w1b   = (bf16*) alloc((size_t)HID * HID * 2);
  bf16* w2b   = (bf16*) alloc((size_t)DPAD * HID * 2);   // 320 rows (grid.y=5 coverage)

  dim3 blk(256);
  convert_x_kernel<<<dim3((MROWS * DPAD) / 256), blk, 0, stream>>>(input, xbf);
  convert_w_kernel<<<dim3((G4 * DPAD + 255) / 256), blk, 0, stream>>>(wih[0], wih0b, G4, DIN, G4, DPAD);
  convert_w_kernel<<<dim3((G4 * HID + 255) / 256),  blk, 0, stream>>>(wih[1], wih1b, G4, HID, G4, HID);
  convert_w_kernel<<<dim3((G4 * HID + 255) / 256),  blk, 0, stream>>>(wih[2], wih2b, G4, HID, G4, HID);
  convert_w_kernel<<<dim3((HID * HID + 255) / 256), blk, 0, stream>>>(w1, w1b, HID, HID, HID, HID);
  convert_w_kernel<<<dim3((DPAD * HID + 255) / 256),blk, 0, stream>>>(w2, w2b, DIN, HID, DPAD, HID);

  dim3 gXP(MROWS / 64, G4 / 64);
  dim3 gLSTM(BATCH / 32);
  // layer 0
  gemm_bf16_wmma<<<gXP, blk, 0, stream>>>(xbf, wih0b, bih[0], bhh[0], xpb, nullptr,
                                          DPAD, G4, G4, 0);
  lstm_layer_kernel<<<gLSTM, blk, 0, stream>>>(xpb, whh[0], h0, c0, actA, hn, cn);
  // layer 1
  gemm_bf16_wmma<<<gXP, blk, 0, stream>>>(actA, wih1b, bih[1], bhh[1], xpb, nullptr,
                                          HID, G4, G4, 0);
  lstm_layer_kernel<<<gLSTM, blk, 0, stream>>>(xpb, whh[1],
      h0 + (size_t)BATCH * HID, c0 + (size_t)BATCH * HID, actB,
      hn + (size_t)BATCH * HID, cn + (size_t)BATCH * HID);
  // layer 2
  gemm_bf16_wmma<<<gXP, blk, 0, stream>>>(actB, wih2b, bih[2], bhh[2], xpb, nullptr,
                                          HID, G4, G4, 0);
  lstm_layer_kernel<<<gLSTM, blk, 0, stream>>>(xpb, whh[2],
      h0 + 2 * (size_t)BATCH * HID, c0 + 2 * (size_t)BATCH * HID, actA,
      hn + 2 * (size_t)BATCH * HID, cn + 2 * (size_t)BATCH * HID);
  // MLP head
  gemm_bf16_wmma<<<dim3(MROWS / 64, 1), blk, 0, stream>>>(actA, w1b, b1, nullptr,
                                          nullptr, actB, HID, HID, HID, /*tanh=*/1);
  gemm_bf16_wmma<<<dim3(MROWS / 64, 5), blk, 0, stream>>>(actB, w2b, b2, nullptr,
                                          out, nullptr, HID, DIN, DIN, 0);
}
